// Encoder_43198781063175
// MI455X (gfx1250) — matrix operand order repaired and checked by reading, not compiled, not run on hardware
//
#include <hip/hip_runtime.h>

#define BB 2
#define LL 2048
#define DD 512
#define HH 8
#define DHH 64
#define NLAYER 3
#define TT (BB * LL)
#define DFF (2 * DD)
#define NCHUNK (LL / 32)

typedef _Float16 f16;
typedef __attribute__((ext_vector_type(16))) _Float16 v16h;
typedef __attribute__((ext_vector_type(8)))  _Float16 v8h;
typedef __attribute__((ext_vector_type(8)))  float    v8f;

// ---------------------------------------------------------------------------
// WMMA helper: D = A(16x32 f16) * B(32x16 f16) + C(16x16 f32)
// ---------------------------------------------------------------------------
__device__ __forceinline__ v8f wmma_f16(v16h a, v16h b, v8f c) {
  return __builtin_amdgcn_wmma_f32_16x16x32_f16(
      /*neg_a=*/false, a, /*neg_b=*/false, b,
      /*c_mod=*/(short)0, c, /*reuse_a=*/false, /*reuse_b=*/false);
}

// A-fragment (16x32 f16, row-major source, row stride = K):
// lane m = lane%16, khalf = lane>>4; halfword j -> K = k0+khalf*16+j
__device__ __forceinline__ v16h load_afrag(const f16* __restrict__ row, int k0, int lh) {
  return *(const v16h*)(row + k0 + lh * 16);
}

// B-fragment (32x16 f16) from a [N,K] row-major (pre-transposed) matrix:
// lane n = lane%16, kgroup = lane>>4; halfword h -> K = k0 + kgroup*16 + h
__device__ __forceinline__ v16h load_bfrag(const f16* __restrict__ row, int k0, int lh) {
  return *(const v16h*)(row + k0 + lh * 16);
}

// ---------------------------------------------------------------------------
// CDNA5 async global->LDS copy (16B per lane) + ASYNCcnt wait
// Builtin signature (from compiler diagnostic): param1 = v4i in AS1 (__device__)
// ---------------------------------------------------------------------------
typedef int v4i_g __attribute__((vector_size(16)));

__device__ __forceinline__ void async_cp16(const void* g, void* l) {
#if __has_builtin(__builtin_amdgcn_global_load_async_to_lds_b128)
  __builtin_amdgcn_global_load_async_to_lds_b128(
      (__attribute__((address_space(1))) v4i_g*)(g),
      (__attribute__((address_space(3))) v4i_g*)(l), 0, 0);
#else
  unsigned int loff = (unsigned int)(unsigned long long)
      (__attribute__((address_space(3))) void*)(l);
  asm volatile("global_load_async_to_lds_b128 %0, %1, off"
               :: "v"(loff), "v"(g) : "memory");
#endif
}

template <int N>
__device__ __forceinline__ void wait_async() {
#if __has_builtin(__builtin_amdgcn_s_wait_asynccnt)
  __builtin_amdgcn_s_wait_asynccnt(N);
#else
  asm volatile("s_wait_asynccnt %0" :: "i"(N) : "memory");
#endif
}

// ---------------------------------------------------------------------------
// Generic WMMA GEMM: out[M,N] = A[M,K](f16) * BT[N,K](f16)^T + bias
// 2x2 register blocking: one 32x32 tile per wave (4 WMMAs per 32-deep k-step),
// 8 waves per 256-thread block. Optional ReLU / f32 residual / f16|f32 out.
// ---------------------------------------------------------------------------
__global__ void gemm_wmma(const f16* __restrict__ A, const f16* __restrict__ BT,
                          const float* __restrict__ bias,
                          f16* __restrict__ outH, float* __restrict__ outF,
                          const float* __restrict__ resid,
                          int M, int N, int K, int relu) {
  int wave = (blockIdx.x * blockDim.x + threadIdx.x) >> 5;
  int lane = threadIdx.x & 31;
  int lm = lane & 15, lh = lane >> 4;
  int tilesN = N >> 5;                      // 32-wide tiles
  int mt = wave / tilesN, nt = wave % tilesN;
  if (mt * 32 >= M) return;

  const f16* arow0 = A  + (size_t)(mt * 32 + lm) * K;
  const f16* arow1 = arow0 + (size_t)16 * K;
  const f16* brow0 = BT + (size_t)(nt * 32 + lm) * K;
  const f16* brow1 = brow0 + (size_t)16 * K;

  v8f acc00 = {}, acc01 = {}, acc10 = {}, acc11 = {};
  for (int k0 = 0; k0 < K; k0 += 32) {
    v16h a0 = load_afrag(arow0, k0, lh);
    v16h a1 = load_afrag(arow1, k0, lh);
    v16h b0 = load_bfrag(brow0, k0, lh);
    v16h b1 = load_bfrag(brow1, k0, lh);
    acc00 = wmma_f16(a0, b0, acc00);
    acc01 = wmma_f16(a0, b1, acc01);
    acc10 = wmma_f16(a1, b0, acc10);
    acc11 = wmma_f16(a1, b1, acc11);
  }

  int n0 = nt * 32 + lm, n1 = n0 + 16;
  float bv0 = bias ? bias[n0] : 0.0f;
  float bv1 = bias ? bias[n1] : 0.0f;
#pragma unroll
  for (int r = 0; r < 8; r++) {
    int mA = mt * 32 + r + 8 * lh;          // C layout: VGPR r, lane-half lh
    int mB = mA + 16;
    size_t i00 = (size_t)mA * N + n0, i01 = (size_t)mA * N + n1;
    size_t i10 = (size_t)mB * N + n0, i11 = (size_t)mB * N + n1;
    float v00 = acc00[r] + bv0, v01 = acc01[r] + bv1;
    float v10 = acc10[r] + bv0, v11 = acc11[r] + bv1;
    if (relu) {
      v00 = fmaxf(v00, 0.0f); v01 = fmaxf(v01, 0.0f);
      v10 = fmaxf(v10, 0.0f); v11 = fmaxf(v11, 0.0f);
    }
    if (resid) { v00 += resid[i00]; v01 += resid[i01]; v10 += resid[i10]; v11 += resid[i11]; }
    if (outF)  { outF[i00] = v00; outF[i01] = v01; outF[i10] = v10; outF[i11] = v11; }
    if (outH)  { outH[i00] = (f16)v00; outH[i01] = (f16)v01;
                 outH[i10] = (f16)v10; outH[i11] = (f16)v11; }
  }
}

// ---------------------------------------------------------------------------
// LayerNorm (f32 in -> f16 out), one wave per row of D=512
// ---------------------------------------------------------------------------
__global__ void layernorm_f16(const float* __restrict__ x, const float* __restrict__ g,
                              const float* __restrict__ b, f16* __restrict__ out) {
  int lane = threadIdx.x & 31;
  int row  = blockIdx.x * (blockDim.x >> 5) + (threadIdx.x >> 5);
  const float* xr = x + (size_t)row * DD;
  float s = 0.0f, s2 = 0.0f;
#pragma unroll
  for (int i = lane; i < DD; i += 32) { float v = xr[i]; s += v; s2 += v * v; }
#pragma unroll
  for (int off = 16; off > 0; off >>= 1) {
    s  += __shfl_xor(s,  off);
    s2 += __shfl_xor(s2, off);
  }
  float mean = s * (1.0f / DD);
  float var  = s2 * (1.0f / DD) - mean * mean;
  float rs   = rsqrtf(var + 1e-5f);
  f16* orow = out + (size_t)row * DD;
#pragma unroll
  for (int i = lane; i < DD; i += 32)
    orow[i] = (f16)((xr[i] - mean) * rs * g[i] + b[i]);
}

// ---------------------------------------------------------------------------
// Weight convert + transpose:  w[K,N] f32  ->  wT[N,K] f16
// ---------------------------------------------------------------------------
__global__ void convert_wT(const float* __restrict__ w, f16* __restrict__ wT,
                           int K, int N) {
  int e = blockIdx.x * blockDim.x + threadIdx.x;
  if (e >= K * N) return;
  int k = e / N, n = e % N;
  wT[(size_t)n * K + k] = (f16)w[e];
}

// V: [B*L, D] f16 -> VT: [B, H, DH, L] f16 (so P*V fragments are contiguous)
__global__ void transpose_v16(const f16* __restrict__ v, f16* __restrict__ vT) {
  int e = blockIdx.x * blockDim.x + threadIdx.x;
  if (e >= TT * DD) return;
  int tok = e / DD, c = e % DD;
  int hh = c / DHH, d = c % DHH;
  int b = tok / LL, l = tok % LL;
  vT[(((size_t)b * HH + hh) * DHH + d) * LL + l] = v[e];
}

// ---------------------------------------------------------------------------
// Flash attention, async-pipelined:
//   wave = 16 queries of one (b,h); K/V chunks (32 keys) double-buffered in LDS
//   via global_load_async_to_lds_b128; s_wait_asynccnt 16 overlaps the next
//   chunk's DMA with this chunk's 8 WMMAs + online softmax.
// ---------------------------------------------------------------------------
__device__ __forceinline__ void issue_chunk(const f16* kbase, const f16* vbase, int kc,
                                            _Float16* dK, _Float16* dV, int lane) {
  // K tile: 32 keys x 64 halfs (row-major, ld=64) ; 256 x 16B segments
#pragma unroll
  for (int j = 0; j < 8; j++) {
    int seg = j * 32 + lane;
    int row = seg >> 3, s = seg & 7;
    async_cp16(kbase + (size_t)(kc * 32 + row) * DD + s * 8, dK + row * 64 + s * 8);
  }
  // V tile: 64 d-rows x 32 keys (row-major, ld=32) from VT[.., d, L]
#pragma unroll
  for (int j = 0; j < 8; j++) {
    int seg = j * 32 + lane;
    int d = seg >> 2, s = seg & 3;
    async_cp16(vbase + (size_t)d * LL + kc * 32 + s * 8, dV + d * 32 + s * 8);
  }
}

__global__ void attention_wmma(const f16* __restrict__ Q, const f16* __restrict__ Kc,
                               const f16* __restrict__ VT,
                               const unsigned char* __restrict__ mask,
                               f16* __restrict__ ctx) {
  __shared__ __align__(64) _Float16 ldsK[8][2][32 * 64];
  __shared__ __align__(64) _Float16 ldsV[8][2][64 * 32];
  __shared__ __align__(64) _Float16 plds[8][16 * 32];

  int wv   = threadIdx.x >> 5;
  int wave = blockIdx.x * 8 + wv;
  int lane = threadIdx.x & 31;
  int lm = lane & 15, lh = lane >> 4;
  int qt = wave & 127;          // L/16 = 128 query tiles
  int bh = wave >> 7;
  int b  = bh >> 3, hh = bh & 7;

  const f16* qrow  = Q  + (size_t)(b * LL + qt * 16 + lm) * DD + hh * DHH;
  const f16* kbase = Kc + (size_t)b * LL * DD + hh * DHH;
  const f16* vbase = VT + (size_t)(b * HH + hh) * DHH * LL;

  v16h aq0 = load_afrag(qrow, 0, lh);    // d = 0..31
  v16h aq1 = load_afrag(qrow, 32, lh);   // d = 32..63

  float rowm[8], rowl[8];
  v8f o0 = {}, o1 = {}, o2 = {}, o3 = {};
#pragma unroll
  for (int r = 0; r < 8; r++) { rowm[r] = -1e30f; rowl[r] = 0.0f; }

  const float scale = 0.125f;  // 1/sqrt(DH=64)

  // prime the pipeline: chunk 0 -> buffer 0
  issue_chunk(kbase, vbase, 0, &ldsK[wv][0][0], &ldsV[wv][0][0], lane);

  for (int kc = 0; kc < NCHUNK; kc++) {
    int cur = kc & 1;
    if (kc + 1 < NCHUNK) {
      issue_chunk(kbase, vbase, kc + 1, &ldsK[wv][cur ^ 1][0], &ldsV[wv][cur ^ 1][0], lane);
      wait_async<16>();   // 16 newest (prefetch) still in flight; current chunk landed
    } else {
      wait_async<0>();
    }
    const _Float16* cK = &ldsK[wv][cur][0];
    const _Float16* cV = &ldsV[wv][cur][0];

    // ---- S = Q * K^T for 16x32 keys (from LDS) ----
    v8f s0 = {}, s1 = {};
    s0 = wmma_f16(aq0, *(const v16h*)(cK + (size_t)lm * 64 + lh * 16), s0);
    s0 = wmma_f16(aq1, *(const v16h*)(cK + (size_t)lm * 64 + 32 + lh * 16), s0);
    s1 = wmma_f16(aq0, *(const v16h*)(cK + (size_t)(16 + lm) * 64 + lh * 16), s1);
    s1 = wmma_f16(aq1, *(const v16h*)(cK + (size_t)(16 + lm) * 64 + 32 + lh * 16), s1);

    // ---- scale + mask (mask indexed by key column = lane) ----
    int key0 = kc * 32 + lm;
    bool m0 = mask[(size_t)b * LL + key0] != 0;
    bool m1 = mask[(size_t)b * LL + key0 + 16] != 0;
#pragma unroll
    for (int r = 0; r < 8; r++) {
      s0[r] = m0 ? -1e30f : s0[r] * scale;
      s1[r] = m1 ? -1e30f : s1[r] * scale;
    }

    // ---- online softmax; rows on (VGPR r, lane-half lh), cols on lanes ----
    float p0[8], p1[8], alpha[8];
#pragma unroll
    for (int r = 0; r < 8; r++) {
      float cm = fmaxf(s0[r], s1[r]);
#pragma unroll
      for (int off = 1; off < 16; off <<= 1) cm = fmaxf(cm, __shfl_xor(cm, off));
      float nm = fmaxf(rowm[r], cm);
      alpha[r] = __expf(rowm[r] - nm);
      rowm[r]  = nm;
      p0[r] = __expf(s0[r] - nm);
      p1[r] = __expf(s1[r] - nm);
      float rsum = p0[r] + p1[r];
#pragma unroll
      for (int off = 1; off < 16; off <<= 1) rsum += __shfl_xor(rsum, off);
      rowl[r] = rowl[r] * alpha[r] + rsum;
    }
#pragma unroll
    for (int r = 0; r < 8; r++) {
      o0[r] *= alpha[r]; o1[r] *= alpha[r]; o2[r] *= alpha[r]; o3[r] *= alpha[r];
    }

    // ---- P: C-layout regs -> LDS (row-major 16x32) -> A-fragment ----
    __syncthreads();
#pragma unroll
    for (int r = 0; r < 8; r++) {
      int m = r + 8 * lh;
      plds[wv][m * 32 + lm]      = (f16)p0[r];
      plds[wv][m * 32 + 16 + lm] = (f16)p1[r];
    }
    __syncthreads();
    v16h ap;
    {
      const _Float16* pr = &plds[wv][lm * 32 + lh * 16];
#pragma unroll
      for (int j = 0; j < 16; j++) { ap[j] = pr[j]; }
    }

    // ---- O += P * V (V tile from LDS, [d][key] ld=32) ----
    o0 = wmma_f16(ap, *(const v16h*)(cV + (size_t)(lm)      * 32 + lh * 16), o0);
    o1 = wmma_f16(ap, *(const v16h*)(cV + (size_t)(16 + lm) * 32 + lh * 16), o1);
    o2 = wmma_f16(ap, *(const v16h*)(cV + (size_t)(32 + lm) * 32 + lh * 16), o2);
    o3 = wmma_f16(ap, *(const v16h*)(cV + (size_t)(48 + lm) * 32 + lh * 16), o3);
  }

  // ---- normalize and write ctx (f16, [B*L, D]) ----
  f16* cbase = ctx + (size_t)(b * LL + qt * 16) * DD + hh * DHH;
#pragma unroll
  for (int r = 0; r < 8; r++) {
    float inv = 1.0f / rowl[r];
    f16* cr = cbase + (size_t)(r + 8 * lh) * DD;
    cr[lm]      = (f16)(o0[r] * inv);
    cr[16 + lm] = (f16)(o1[r] * inv);
    cr[32 + lm] = (f16)(o2[r] * inv);
    cr[48 + lm] = (f16)(o3[r] * inv);
  }
}

// ---------------------------------------------------------------------------
// Host-side orchestration
// ---------------------------------------------------------------------------
extern "C" void kernel_launch(void* const* d_in, const int* in_sizes, int n_in,
                              void* d_out, int out_size, void* d_ws, size_t ws_size,
                              hipStream_t stream) {
  (void)in_sizes; (void)n_in; (void)out_size; (void)ws_size;
  const float* x  = (const float*)d_in[0];
  const unsigned char* mask = (const unsigned char*)d_in[1];
  const float* ln_ag = (const float*)d_in[2];
  const float* ln_ab = (const float*)d_in[3];
  const float* wq = (const float*)d_in[4];  const float* bq = (const float*)d_in[5];
  const float* wk = (const float*)d_in[6];  const float* bk = (const float*)d_in[7];
  const float* wv = (const float*)d_in[8];  const float* bv = (const float*)d_in[9];
  const float* wo = (const float*)d_in[10]; const float* bo = (const float*)d_in[11];
  const float* ln_fg = (const float*)d_in[12];
  const float* ln_fb = (const float*)d_in[13];
  const float* w1 = (const float*)d_in[14]; const float* b1 = (const float*)d_in[15];
  const float* w2 = (const float*)d_in[16]; const float* b2 = (const float*)d_in[17];
  float* out = (float*)d_out;

  char* ws = (char*)d_ws;
  size_t off = 0;
  auto alloc = [&](size_t bytes) -> void* {
    void* p = ws + off;
    off += (bytes + 255) & ~(size_t)255;
    return p;
  };
  float* xbuf = (float*)alloc((size_t)TT * DD * 4);
  f16* h16  = (f16*)alloc((size_t)TT * DD * 2);
  f16* q16  = (f16*)alloc((size_t)TT * DD * 2);
  f16* k16  = (f16*)alloc((size_t)TT * DD * 2);
  f16* v16  = (f16*)alloc((size_t)TT * DD * 2);
  f16* vT16 = (f16*)alloc((size_t)TT * DD * 2);
  f16* c16  = (f16*)alloc((size_t)TT * DD * 2);
  f16* ff16 = (f16*)alloc((size_t)TT * DFF * 2);
  f16* wqT = (f16*)alloc((size_t)NLAYER * DD * DD * 2);
  f16* wkT = (f16*)alloc((size_t)NLAYER * DD * DD * 2);
  f16* wvT = (f16*)alloc((size_t)NLAYER * DD * DD * 2);
  f16* woT = (f16*)alloc((size_t)NLAYER * DD * DD * 2);
  f16* w1T = (f16*)alloc((size_t)NLAYER * DD * DFF * 2);
  f16* w2T = (f16*)alloc((size_t)NLAYER * DFF * DD * 2);

  (void)hipMemcpyAsync(xbuf, x, (size_t)TT * DD * 4, hipMemcpyDeviceToDevice, stream);

  dim3 blk(256);
  int cvtD  = (DD * DD + 255) / 256;
  int cvtFF = (DD * DFF + 255) / 256;
  for (int i = 0; i < NLAYER; i++) {
    size_t oD = (size_t)i * DD * DD, oF = (size_t)i * DD * DFF;
    convert_wT<<<cvtD, blk, 0, stream>>>(wq + oD, wqT + oD, DD, DD);
    convert_wT<<<cvtD, blk, 0, stream>>>(wk + oD, wkT + oD, DD, DD);
    convert_wT<<<cvtD, blk, 0, stream>>>(wv + oD, wvT + oD, DD, DD);
    convert_wT<<<cvtD, blk, 0, stream>>>(wo + oD, woT + oD, DD, DD);
    convert_wT<<<cvtFF, blk, 0, stream>>>(w1 + oF, w1T + oF, DD, DFF);
    convert_wT<<<cvtFF, blk, 0, stream>>>(w2 + oF, w2T + oF, DFF, DD);
  }

  int lnBlocks     = TT / 8;                            // wave per row
  int gemmBlocksD  = (TT / 32) * (DD / 32) / 8;         // 256
  int gemmBlocksFF = (TT / 32) * (DFF / 32) / 8;        // 512
  int tvBlocks     = (TT * DD) / 256;
  int attBlocks    = (BB * HH * (LL / 16)) / 8;         // 256

  for (int i = 0; i < NLAYER; i++) {
    size_t oD = (size_t)i * DD * DD, oF = (size_t)i * DD * DFF;
    layernorm_f16<<<lnBlocks, blk, 0, stream>>>(xbuf, ln_ag + i * DD, ln_ab + i * DD, h16);
    gemm_wmma<<<gemmBlocksD, blk, 0, stream>>>(h16, wqT + oD, bq + i * DD, q16, nullptr, nullptr, TT, DD, DD, 0);
    gemm_wmma<<<gemmBlocksD, blk, 0, stream>>>(h16, wkT + oD, bk + i * DD, k16, nullptr, nullptr, TT, DD, DD, 0);
    gemm_wmma<<<gemmBlocksD, blk, 0, stream>>>(h16, wvT + oD, bv + i * DD, v16, nullptr, nullptr, TT, DD, DD, 0);
    transpose_v16<<<tvBlocks, blk, 0, stream>>>(v16, vT16);
    attention_wmma<<<attBlocks, blk, 0, stream>>>(q16, k16, vT16, mask, c16);
    gemm_wmma<<<gemmBlocksD, blk, 0, stream>>>(c16, woT + oD, bo + i * DD, nullptr, xbuf, xbuf, TT, DD, DD, 0);
    layernorm_f16<<<lnBlocks, blk, 0, stream>>>(xbuf, ln_fg + i * DD, ln_fb + i * DD, h16);
    gemm_wmma<<<gemmBlocksFF, blk, 0, stream>>>(h16, w1T + oF, b1 + i * DFF, ff16, nullptr, nullptr, TT, DFF, DD, 1);
    float* dst = (i == NLAYER - 1) ? out : xbuf;
    gemm_wmma<<<gemmBlocksD, blk, 0, stream>>>(ff16, w2T + oF, b2 + i * DD, nullptr, dst, xbuf, TT, DD, DFF, 0);
  }
}
